// DistanceModule_47794396070409
// MI455X (gfx1250) — matrix operand order repaired and checked by reading, not compiled, not run on hardware
//
#include <hip/hip_runtime.h>
#include <hip/hip_bf16.h>

// ---------------------------------------------------------------------------
// Cosine similarity: out[n,m] = (z[n,:]/||z[n]||) . (cm[m,:]/||cm[m]||)
// n=32768, m=1001, d=512.
//   kernel 1: z row inverse norms (f32, ws) -> applied in GEMM epilogue
//   kernel 2: cm normalized -> f16, padded to 1024 rows (ws)
//   kernel 3: WMMA GEMM, LDS-staged double-buffered tiles:
//             A: coalesced f32 loads -> cvt f16 -> ds_store (cvt co-executes
//                with the XDL WMMA pipe)
//             B: GLOBAL_LOAD_ASYNC_TO_LDS_B128 (ASYNCcnt path), with fallback
//   compute(): software-pipelined B fragment loads (bc/bn ping-pong) so each
//   ds_load has a WMMA-pair of slack instead of a full s_wait_dscnt 0 drain.
// Workspace: n*4 + 1024*512*2 bytes (~1.2 MB)
// ---------------------------------------------------------------------------

typedef __attribute__((ext_vector_type(16))) _Float16 v16h;
typedef __attribute__((ext_vector_type(8)))  _Float16 v8h;
typedef __attribute__((ext_vector_type(8)))  float    v8f;
typedef __attribute__((ext_vector_type(4)))  float    v4f;
typedef __attribute__((ext_vector_type(4)))  int      v4i;

#define DDIM 512
#define EPSN 1e-8f
#define LSTR 72   // LDS row stride in halves: 144B = 36 banks, conflict-free frag reads

#if defined(__has_builtin)
#  if __has_builtin(__builtin_amdgcn_global_load_async_to_lds_b128)
#    define USE_ASYNC_LDS 1
#  endif
#endif
#ifndef USE_ASYNC_LDS
#  define USE_ASYNC_LDS 0
#endif

// HIP pretty-print: AS1 == __device__ (global), AS3 == __shared__ (LDS)
typedef __attribute__((address_space(1))) v4i as1_v4i;
typedef __attribute__((address_space(3))) v4i as3_v4i;

__device__ inline void wait_async_all() {
#if USE_ASYNC_LDS
#  if __has_builtin(__builtin_amdgcn_s_wait_asynccnt)
  __builtin_amdgcn_s_wait_asynccnt(0);
#  else
  asm volatile("s_wait_asynccnt 0" ::: "memory");
#  endif
#endif
}

// ---------------- wave reduction (wave32) ----------------
__device__ inline float wave_reduce_sum(float s) {
#pragma unroll
  for (int o = 16; o > 0; o >>= 1) s += __shfl_xor(s, o, 32);
  return s;
}

// ---------------- kernel 1: z inverse row norms ----------------
__global__ void zinv_kernel(const float* __restrict__ z,
                            float* __restrict__ invn) {
  const int lane = threadIdx.x & 31;
  const int row  = blockIdx.x * 8 + (threadIdx.x >> 5);
  const float* zr = z + (size_t)row * DDIM;
  const int base = lane * 16;
  v4f f0 = *(const v4f*)(zr + base + 0);
  v4f f1 = *(const v4f*)(zr + base + 4);
  v4f f2 = *(const v4f*)(zr + base + 8);
  v4f f3 = *(const v4f*)(zr + base + 12);
  float s = 0.f;
#pragma unroll
  for (int i = 0; i < 4; ++i)
    s += f0[i]*f0[i] + f1[i]*f1[i] + f2[i]*f2[i] + f3[i]*f3[i];
  s = wave_reduce_sum(s);
  if (lane == 0) invn[row] = 1.0f / fmaxf(sqrtf(s), EPSN);
}

// ---------------- kernel 2: normalize cm -> f16, pad with zeros ----------------
__global__ void cnorm_kernel(const float* __restrict__ cm,
                             _Float16* __restrict__ cn, int m) {
  const int lane = threadIdx.x & 31;
  const int row  = blockIdx.x * 8 + (threadIdx.x >> 5);
  const int base = lane * 16;
  v8h h0 = (v8h)(_Float16)0.0f;
  v8h h1 = (v8h)(_Float16)0.0f;
  if (row < m) {
    const float* cr = cm + (size_t)row * DDIM;
    v4f f0 = *(const v4f*)(cr + base + 0);
    v4f f1 = *(const v4f*)(cr + base + 4);
    v4f f2 = *(const v4f*)(cr + base + 8);
    v4f f3 = *(const v4f*)(cr + base + 12);
    float s = 0.f;
#pragma unroll
    for (int i = 0; i < 4; ++i)
      s += f0[i]*f0[i] + f1[i]*f1[i] + f2[i]*f2[i] + f3[i]*f3[i];
    s = wave_reduce_sum(s);
    const float sc = 1.0f / fmaxf(sqrtf(s), EPSN);
#pragma unroll
    for (int i = 0; i < 4; ++i) {
      h0[i]     = (_Float16)(f0[i] * sc);
      h0[4 + i] = (_Float16)(f1[i] * sc);
      h1[i]     = (_Float16)(f2[i] * sc);
      h1[4 + i] = (_Float16)(f3[i] * sc);
    }
  }
  _Float16* dst = cn + (size_t)row * DDIM + base;
  *(v8h*)(dst + 0) = h0;
  *(v8h*)(dst + 8) = h1;
}

__device__ inline v8f wmma_f16(v16h a, v16h b, v8f c) {
  return __builtin_amdgcn_wmma_f32_16x16x32_f16(
      false, a, false, b, (short)0, c, false, false);
}

__device__ inline v16h cat8(v8h lo, v8h hi) {
  return __builtin_shufflevector(lo, hi, 0, 1, 2, 3, 4, 5, 6, 7,
                                 8, 9, 10, 11, 12, 13, 14, 15);
}

// ---------------- kernel 3: WMMA GEMM ----------------
// Block 256 threads = 8 waves; block tile 128(M) x 128(N), BK=64, 2x buffered.
// Wave grid 4(M) x 2(N); each wave computes 32x64 = 2x4 tiles of 16x16.
__global__ void __launch_bounds__(256)
gemm_kernel(const float* __restrict__ z, const _Float16* __restrict__ cn,
            const float* __restrict__ invn, float* __restrict__ out, int m) {
  const int tid   = threadIdx.x;
  const int lane  = tid & 31;
  const int laneM = lane & 15;
  const int hiA   = (lane & 16) ? 8  : 0;  // A frag K half-offset (ISA A 16x32 f16 layout)
  const int hiB   = (lane & 16) ? 8  : 0;  // B frag K half-offset
  const int wid   = tid >> 5;
  const int waveM = wid >> 1;   // 0..3
  const int waveN = wid & 1;    // 0..1

  __shared__ __align__(16) _Float16 smA[2][128 * LSTR];
  __shared__ __align__(16) _Float16 smB[2][128 * LSTR];

  // Staging map: pass j covers rows [j*32,(j+1)*32); thread -> (row, 8-elem col)
  const int sR = tid >> 3;        // 0..31
  const int sC = (tid & 7) * 8;   // 0,8,...,56
  const float*    gA0 = z  + (size_t)(blockIdx.x * 128 + sR) * DDIM + sC;
  const _Float16* gB0 = cn + (size_t)(blockIdx.y * 128 + sR) * DDIM + sC;

  auto stage = [&](int buf, int k0) {
#pragma unroll
    for (int j = 0; j < 4; ++j) {
      const int row = j * 32 + sR;
      // ---- A: coalesced f32 load, convert, ds_store ----
      const float* ga = gA0 + (size_t)(j * 32) * DDIM + k0;
      v4f x0 = *(const v4f*)(ga + 0);
      v4f x1 = *(const v4f*)(ga + 4);
      v8h h;
#pragma unroll
      for (int i = 0; i < 4; ++i) {
        h[i]     = (_Float16)x0[i];
        h[4 + i] = (_Float16)x1[i];
      }
      *(v8h*)&smA[buf][row * LSTR + sC] = h;
      // ---- B: f16, async global->LDS (falls back to load+ds_store) ----
      const _Float16* gb = gB0 + (size_t)(j * 32) * DDIM + k0;
      _Float16* lb = &smB[buf][row * LSTR + sC];
#if USE_ASYNC_LDS
      __builtin_amdgcn_global_load_async_to_lds_b128((as1_v4i*)gb, (as3_v4i*)lb, 0, 0);
#else
      *(v8h*)lb = *(const v8h*)gb;
#endif
    }
  };

  v8f acc[2][4] = {};

  auto loadA = [&](int buf, int kk, int mi) -> v16h {
    const _Float16* p = &smA[buf][(waveM * 32 + mi * 16 + laneM) * LSTR + kk + hiA];
    return cat8(*(const v8h*)p, *(const v8h*)(p + 16));
  };
  auto loadB = [&](int buf, int kk, int ni) -> v16h {
    const _Float16* p = &smB[buf][(waveN * 64 + ni * 16 + laneM) * LSTR + kk + hiB];
    return cat8(*(const v8h*)p, *(const v8h*)(p + 16));
  };

  // 2 k-steps (kk=0,32) x 4 N-tiles, software-pipelined B loads.
  auto compute = [&](int buf) {
    v16h a0 = loadA(buf, 0, 0);
    v16h a1 = loadA(buf, 0, 1);
    v16h bc = loadB(buf, 0, 0);
#pragma unroll
    for (int s = 0; s < 8; ++s) {
      const int ni = s & 3;
      v16h bn;
      if (s < 7) bn = loadB(buf, ((s + 1) >> 2) * 32, (s + 1) & 3);
      acc[0][ni] = wmma_f16(a0, bc, acc[0][ni]);
      acc[1][ni] = wmma_f16(a1, bc, acc[1][ni]);
      if (s == 2) {  // preload kk=32 A fragments with plenty of slack
        a0 = loadA(buf, 32, 0);
        a1 = loadA(buf, 32, 1);
      }
      if (s < 7) bc = bn;
    }
  };

  stage(0, 0);
#pragma unroll 1
  for (int c = 0; c < 8; ++c) {
    wait_async_all();
    __syncthreads();
    if (c < 7) stage((c + 1) & 1, (c + 1) * 64);
    compute(c & 1);
  }

  // Epilogue: C/D layout -> element (M = r + 8*(lane>=16), N = lane&15).
  const int rowBase = blockIdx.x * 128 + waveM * 32;
  const int colBase = blockIdx.y * 128 + waveN * 64;
  const int laneHi  = (lane >> 4) & 1;
#pragma unroll
  for (int mi = 0; mi < 2; ++mi) {
    const int rbase = rowBase + mi * 16 + laneHi * 8;
    float inv[8];
#pragma unroll
    for (int r = 0; r < 8; ++r) inv[r] = invn[rbase + r];
#pragma unroll
    for (int ni = 0; ni < 4; ++ni) {
      const int col = colBase + ni * 16 + laneM;
      if (col < m) {
#pragma unroll
        for (int r = 0; r < 8; ++r) {
          out[(size_t)(rbase + r) * (size_t)m + col] = acc[mi][ni][r] * inv[r];
        }
      }
    }
  }
}

// ---------------------------------------------------------------------------
extern "C" void kernel_launch(void* const* d_in, const int* in_sizes, int n_in,
                              void* d_out, int out_size, void* d_ws, size_t ws_size,
                              hipStream_t stream) {
  const float* z  = (const float*)d_in[0];
  const float* cm = (const float*)d_in[1];
  const int n = in_sizes[0] / DDIM;   // 32768
  const int m = in_sizes[1] / DDIM;   // 1001
  const int mPad = ((m + 127) / 128) * 128;  // 1024

  float*    invn = (float*)d_ws;
  size_t    off  = (((size_t)n * sizeof(float)) + 255) & ~(size_t)255;
  _Float16* cn   = (_Float16*)((char*)d_ws + off);

  zinv_kernel<<<n / 8, 256, 0, stream>>>(z, invn);
  cnorm_kernel<<<mPad / 8, 256, 0, stream>>>(cm, cn, m);
  dim3 grid(n / 128, mPad / 128);
  gemm_kernel<<<grid, 256, 0, stream>>>(z, cn, invn, (float*)d_out, m);
}
